// MultiheadAttention_26027501814440
// MI455X (gfx1250) — compile-verified, matrix operand order checked by reading, not run on hardware
//
#include <hip/hip_runtime.h>

// Problem constants (match reference)
#define B_  4
#define S_  2048
#define D_  1024
#define H_  16
#define DK_ 64
#define MTOT (B_ * S_)

typedef __attribute__((ext_vector_type(16))) __bf16 v16bf;
typedef __attribute__((ext_vector_type(8)))  float  v8f;
typedef int gnu_int4 __attribute__((vector_size(16)));  // matches builtin param

union FragBF {
  v16bf v;
  unsigned int u[8];
};

#if defined(__gfx1250__) && __has_builtin(__builtin_amdgcn_global_load_async_to_lds_b128)
#define HAVE_ASYNC_LDS 1
#endif

// round-to-nearest-even f32 -> bf16
__device__ __forceinline__ unsigned short bf16_1(float f) {
  unsigned int a = __float_as_uint(f);
  return (unsigned short)((a + 0x7fffu + ((a >> 16) & 1u)) >> 16);
}
__device__ __forceinline__ unsigned int pk_bf16(float lo, float hi) {
  unsigned int a = __float_as_uint(lo);
  unsigned int b = __float_as_uint(hi);
  a = (a + 0x7fffu + ((a >> 16) & 1u)) >> 16;
  b = (b + 0x7fffu + ((b >> 16) & 1u)) & 0xffff0000u;
  return (a & 0xffffu) | b;
}

// ---------------------------------------------------------------------------
// One-shot f32 -> packed bf16 conversion (bandwidth bound)
// ---------------------------------------------------------------------------
__global__ __launch_bounds__(256) void cvt_f32_bf16(
    const float* __restrict__ X, unsigned short* __restrict__ Y, int n) {
  int i = (blockIdx.x * 256 + threadIdx.x) * 8;
  if (i + 8 <= n) {
    float4 a = *(const float4*)(X + i);
    float4 b = *(const float4*)(X + i + 4);
    uint4 o;
    o.x = pk_bf16(a.x, a.y); o.y = pk_bf16(a.z, a.w);
    o.z = pk_bf16(b.x, b.y); o.w = pk_bf16(b.z, b.w);
    *(uint4*)(Y + i) = o;
  }
}

// ---------------------------------------------------------------------------
// C[M,N] = A[M,K] * W[N,K]^T with A, W in bf16 (native WMMA fragment loads:
// each 16x32 fragment = two global_load_b128, no conversion VALU).
// Block: 128 threads = 4 waves. Wave computes a 16x64 tile (4 WMMA accums).
// ---------------------------------------------------------------------------
template <bool OUT_BF16>
__global__ __launch_bounds__(128) void gemm_xwT(
    const unsigned short* __restrict__ A, const unsigned short* __restrict__ W,
    void* __restrict__ Cout, int M, int N, int K) {
  const int lane = threadIdx.x & 31;
  const int wave = threadIdx.x >> 5;
  const int half = lane >> 4;
  const int l16  = lane & 15;
  const int row0 = blockIdx.y << 4;
  const int col0 = (blockIdx.x * 4 + wave) << 6;

  const unsigned short* arow  = A + (size_t)(row0 + l16) * K;
  const unsigned short* wrow0 = W + (size_t)(col0 + l16) * K;

  v8f acc[4] = {};

  for (int kb = 0; kb < K; kb += 32) {
    FragBF a;
    {
      const unsigned short* ap = arow + kb + half * 8;  // 16B aligned
      uint4 x0 = *(const uint4*)(ap);        // K = kb+half*8   .. +7
      uint4 x1 = *(const uint4*)(ap + 16);   // K = kb+16+half*8.. +7
      a.u[0] = x0.x; a.u[1] = x0.y; a.u[2] = x0.z; a.u[3] = x0.w;
      a.u[4] = x1.x; a.u[5] = x1.y; a.u[6] = x1.z; a.u[7] = x1.w;
    }
    if (kb + 32 < K) __builtin_prefetch(arow + kb + 32, 0, 0);

#pragma unroll
    for (int t = 0; t < 4; ++t) {
      const unsigned short* wp = wrow0 + (size_t)(t << 4) * K + kb + half * 8;
      uint4 y0 = *(const uint4*)(wp);
      uint4 y1 = *(const uint4*)(wp + 16);
      FragBF b;
      b.u[0] = y0.x; b.u[1] = y0.y; b.u[2] = y0.z; b.u[3] = y0.w;
      b.u[4] = y1.x; b.u[5] = y1.y; b.u[6] = y1.z; b.u[7] = y1.w;
      acc[t] = __builtin_amdgcn_wmma_f32_16x16x32_bf16(
          false, a.v, false, b.v, (short)0, acc[t], false, false);
    }
  }

#pragma unroll
  for (int t = 0; t < 4; ++t) {
#pragma unroll
    for (int r = 0; r < 8; ++r) {
      size_t m = (size_t)(row0 + r + 8 * half);
      size_t n = (size_t)(col0 + t * 16 + l16);
      if constexpr (OUT_BF16)
        ((unsigned short*)Cout)[m * N + n] = bf16_1(acc[t][r]);
      else
        ((float*)Cout)[m * N + n] = acc[t][r];
    }
  }
}

// ---------------------------------------------------------------------------
// Causal flash attention over bf16 Q/K/V stored [B,S,D] (head-interleaved).
// Block = 256 threads (8 waves) = 128 query rows of one (b,h).
// K tile staged via GLOBAL_LOAD_ASYNC_TO_LDS_B128 (ASYNCcnt);
// V stored transposed in LDS so B-fragments are contiguous bf16 pairs.
// ---------------------------------------------------------------------------
__global__ __launch_bounds__(256) void attn_causal(
    const unsigned short* __restrict__ Q,
    const unsigned short* __restrict__ K,
    const unsigned short* __restrict__ V,
    unsigned short* __restrict__ Ctx) {
  __shared__ unsigned int   ldsK[64 * 32];       // 64 keys x 64 dk (bf16 pairs)
  __shared__ unsigned short ldsVt[64 * 64];      // transposed: [dk][key]
  __shared__ unsigned short ldsP[8][16 * 64];    // per-wave P tile, row major

  const int bh   = blockIdx.y;
  const int b    = bh >> 4;    // H = 16
  const int h    = bh & 15;
  const int q0   = blockIdx.x << 7;
  const int lane = threadIdx.x & 31;
  const int wave = threadIdx.x >> 5;
  const int half = lane >> 4;
  const int l16  = lane & 15;
  const int qw   = q0 + (wave << 4);

  const size_t bS = (size_t)b * S_;

  // Q fragments: 16 rows x 64 dk -> two 16x32 A-fragments
  FragBF qf[2];
  {
    const unsigned short* qrow = Q + (bS + qw + l16) * D_ + h * DK_;
#pragma unroll
    for (int ks = 0; ks < 2; ++ks)
#pragma unroll
      for (int j = 0; j < 4; ++j) {
        qf[ks].u[j]     = *(const unsigned int*)(qrow + ks * 32 + half * 8 + 2 * j);
        qf[ks].u[j + 4] = *(const unsigned int*)(qrow + ks * 32 + 16 + half * 8 + 2 * j);
      }
  }

  float mrow[8], lrow[8];
  v8f oacc[4] = {};
#pragma unroll
  for (int r = 0; r < 8; ++r) { mrow[r] = -3.0e38f; lrow[r] = 0.0f; }

  const int kEnd = q0 + 128;
  for (int k0 = 0; k0 < kEnd; k0 += 64) {
    __syncthreads();

#ifdef HAVE_ASYNC_LDS
    // K tile: 64 rows x 128B = 512 x 16B chunks, async DMA into LDS
    for (int c = threadIdx.x; c < 512; c += 256) {
      int ky = c >> 3, seg = c & 7;
      const unsigned short* g = K + (bS + k0 + ky) * D_ + h * DK_ + seg * 8;
      __builtin_amdgcn_global_load_async_to_lds_b128(
          (__attribute__((address_space(1))) gnu_int4*)g,
          (__attribute__((address_space(3))) gnu_int4*)&ldsK[ky * 32 + seg * 4],
          0, 0);
    }
    // V tile: load + transpose into LDS
    for (int i = threadIdx.x; i < 64 * 32; i += 256) {
      int ky = i >> 5, dp = i & 31;
      unsigned int vv =
          *(const unsigned int*)(V + (bS + k0 + ky) * D_ + h * DK_ + dp * 2);
      ldsVt[(dp * 2 + 0) * 64 + ky] = (unsigned short)(vv & 0xffffu);
      ldsVt[(dp * 2 + 1) * 64 + ky] = (unsigned short)(vv >> 16);
    }
    asm volatile("s_wait_asynccnt 0" ::: "memory");
#else
    for (int i = threadIdx.x; i < 64 * 32; i += 256) {
      int ky = i >> 5, dp = i & 31;
      size_t g = (bS + k0 + ky) * D_ + h * DK_ + dp * 2;
      ldsK[i] = *(const unsigned int*)(K + g);
      unsigned int vv = *(const unsigned int*)(V + g);
      ldsVt[(dp * 2 + 0) * 64 + ky] = (unsigned short)(vv & 0xffffu);
      ldsVt[(dp * 2 + 1) * 64 + ky] = (unsigned short)(vv >> 16);
    }
#endif
    __syncthreads();

    if (k0 <= qw + 15) {  // causal: skip tiles entirely above the diagonal
      float sv[4][8];
      float tm[8];
#pragma unroll
      for (int r = 0; r < 8; ++r) tm[r] = -3.0e38f;

      // scores: S = Q K^T * 1/sqrt(64), masked
#pragma unroll
      for (int t = 0; t < 4; ++t) {
        v8f s = {};
#pragma unroll
        for (int ks = 0; ks < 2; ++ks) {
          FragBF kf;
          const unsigned int* kr = ldsK + (t * 16 + l16) * 32 + ks * 16 + half * 4;
#pragma unroll
          for (int j = 0; j < 4; ++j) { kf.u[j] = kr[j]; kf.u[j + 4] = kr[8 + j]; }
          s = __builtin_amdgcn_wmma_f32_16x16x32_bf16(
              false, qf[ks].v, false, kf.v, (short)0, s, false, false);
        }
        const int ki = k0 + t * 16 + l16;
#pragma unroll
        for (int r = 0; r < 8; ++r) {
          const int qi = qw + r + 8 * half;
          float x = s[r] * 0.125f;
          if (ki > qi) x = -3.0e38f;
          sv[t][r] = x;
          tm[r] = fmaxf(tm[r], x);
        }
      }
      // row max across the 16-lane group (rows live in one half-wave)
#pragma unroll
      for (int mk = 1; mk < 16; mk <<= 1)
#pragma unroll
        for (int r = 0; r < 8; ++r)
          tm[r] = fmaxf(tm[r], __shfl_xor(tm[r], mk, 32));

      float alpha[8], rs[8];
#pragma unroll
      for (int r = 0; r < 8; ++r) {
        float nm = fmaxf(mrow[r], tm[r]);
        alpha[r] = __expf(mrow[r] - nm);
        mrow[r]  = nm;
        rs[r]    = 0.0f;
      }

      // P = exp(S - m), written to per-wave LDS (C-layout -> A-layout bridge)
      unsigned short* pw = &ldsP[wave][0];
#pragma unroll
      for (int t = 0; t < 4; ++t)
#pragma unroll
        for (int r = 0; r < 8; ++r) {
          float p = __expf(sv[t][r] - mrow[r]);
          rs[r] += p;
          pw[(r + 8 * half) * 64 + t * 16 + l16] = bf16_1(p);
        }
#pragma unroll
      for (int mk = 1; mk < 16; mk <<= 1)
#pragma unroll
        for (int r = 0; r < 8; ++r)
          rs[r] += __shfl_xor(rs[r], mk, 32);
#pragma unroll
      for (int r = 0; r < 8; ++r) lrow[r] = lrow[r] * alpha[r] + rs[r];

      // rescale O, then O += P V
#pragma unroll
      for (int t2 = 0; t2 < 4; ++t2)
#pragma unroll
        for (int r = 0; r < 8; ++r) oacc[t2][r] *= alpha[r];

      asm volatile("s_wait_dscnt 0" ::: "memory");  // P store -> A-frag load

#pragma unroll
      for (int ks = 0; ks < 2; ++ks) {
        FragBF pf;
#pragma unroll
        for (int j = 0; j < 4; ++j) {
          pf.u[j]     = *(const unsigned int*)&pw[l16 * 64 + ks * 32 + half * 8 + 2 * j];
          pf.u[j + 4] = *(const unsigned int*)&pw[l16 * 64 + ks * 32 + 16 + half * 8 + 2 * j];
        }
#pragma unroll
        for (int t2 = 0; t2 < 4; ++t2) {
          FragBF vf;
          const unsigned short* vr = ldsVt + (t2 * 16 + l16) * 64 + ks * 32 + half * 8;
#pragma unroll
          for (int j = 0; j < 4; ++j) {
            vf.u[j]     = *(const unsigned int*)(vr + 2 * j);
            vf.u[j + 4] = *(const unsigned int*)(vr + 16 + 2 * j);
          }
          oacc[t2] = __builtin_amdgcn_wmma_f32_16x16x32_bf16(
              false, pf.v, false, vf.v, (short)0, oacc[t2], false, false);
        }
      }
    }
  }

  // normalize, convert to bf16 ctx in [B,S,D] layout for the output projection
#pragma unroll
  for (int t2 = 0; t2 < 4; ++t2) {
#pragma unroll
    for (int r = 0; r < 8; ++r) {
      float inv = 1.0f / lrow[r];
      Ctx[(bS + qw + r + 8 * half) * D_ + h * DK_ + t2 * 16 + l16] =
          bf16_1(oacc[t2][r] * inv);
    }
  }
}

// ---------------------------------------------------------------------------
extern "C" void kernel_launch(void* const* d_in, const int* in_sizes, int n_in,
                              void* d_out, int out_size, void* d_ws, size_t ws_size,
                              hipStream_t stream) {
  (void)in_sizes; (void)n_in; (void)out_size; (void)ws_size;
  const float* query = (const float*)d_in[0];
  const float* key   = (const float*)d_in[1];
  const float* value = (const float*)d_in[2];
  // d_in[3] = causal mask, baked into the attention kernel
  const float* wq = (const float*)d_in[4];
  const float* wk = (const float*)d_in[5];
  const float* wv = (const float*)d_in[6];
  const float* wo = (const float*)d_in[7];
  float* out = (float*)d_out;

  const size_t NW = (size_t)D_ * D_;     // weight elems
  const size_t NX = (size_t)MTOT * D_;   // activation elems

  // workspace: 4 bf16 weights (8MB) + 3 bf16 inputs (48MB) + 3 bf16 q/k/v
  // (48MB); ctx(bf16) aliases the query-bf16 region after projections finish.
  unsigned short* wqb = (unsigned short*)d_ws;
  unsigned short* wkb = wqb + NW;
  unsigned short* wvb = wkb + NW;
  unsigned short* wob = wvb + NW;
  unsigned short* xq  = wob + NW;
  unsigned short* xk  = xq + NX;
  unsigned short* xv  = xk + NX;
  unsigned short* qb  = xv + NX;
  unsigned short* kb  = qb + NX;
  unsigned short* vb  = kb + NX;
  unsigned short* ctx = xq;  // reuse: xq dead after first projection GEMM

  // 1) one-shot conversions to bf16
  cvt_f32_bf16<<<dim3(NW / 2048), dim3(256), 0, stream>>>(wq, wqb, (int)NW);
  cvt_f32_bf16<<<dim3(NW / 2048), dim3(256), 0, stream>>>(wk, wkb, (int)NW);
  cvt_f32_bf16<<<dim3(NW / 2048), dim3(256), 0, stream>>>(wv, wvb, (int)NW);
  cvt_f32_bf16<<<dim3(NW / 2048), dim3(256), 0, stream>>>(wo, wob, (int)NW);
  cvt_f32_bf16<<<dim3(NX / 2048), dim3(256), 0, stream>>>(query, xq, (int)NX);
  cvt_f32_bf16<<<dim3(NX / 2048), dim3(256), 0, stream>>>(key,   xk, (int)NX);
  cvt_f32_bf16<<<dim3(NX / 2048), dim3(256), 0, stream>>>(value, xv, (int)NX);

  // 2) projections (bf16 x bf16 -> bf16)
  dim3 gg(D_ / 256, MTOT / 16);
  dim3 bb(128);
  gemm_xwT<true><<<gg, bb, 0, stream>>>(xq, wqb, qb, MTOT, D_, D_);
  gemm_xwT<true><<<gg, bb, 0, stream>>>(xk, wkb, kb, MTOT, D_, D_);
  gemm_xwT<true><<<gg, bb, 0, stream>>>(xv, wvb, vb, MTOT, D_, D_);

  // 3) causal flash attention -> bf16 ctx
  dim3 ga(S_ / 128, B_ * H_);
  attn_causal<<<ga, dim3(256), 0, stream>>>(qb, kb, vb, ctx);

  // 4) output projection -> f32 result
  gemm_xwT<false><<<gg, bb, 0, stream>>>(ctx, wob, out, MTOT, D_, D_);
}
